// semantic_align_67662914781840
// MI455X (gfx1250) — matrix operand order repaired and checked by reading, not compiled, not run on hardware
//
#include <hip/hip_runtime.h>
#include <hip/hip_bf16.h>

typedef __attribute__((ext_vector_type(16))) __bf16 v16bf;
typedef __attribute__((ext_vector_type(8)))  float  v8f;

#define BATCH   16
#define LFEAT   2048
#define L1Q     4096
#define DIMC    256
#define NT_TOTAL (BATCH * L1Q)
#define TILE_M  32
#define LSTR    264   // LDS row stride in halves (16B-aligned, bank-staggered)
#define HSTR    72    // hidden (64-wide) row stride in halves

// packed bf16 weight-fragment regions in workspace (offsets in halves)
#define OFF_C1W1 0
#define OFF_C1W2 16384
#define OFF_C2W1 32768
#define OFF_C2W2 49152
#define OFF_GW1  65536
#define WS_HALVES 196608

union V16 { v16bf v; uint4 u[2]; };
union H8  { uint4 u; __bf16 h[8]; };

static __device__ __forceinline__ v8f wmma_bf16(v16bf a, v16bf b, v8f c) {
    // emits v_wmma_f32_16x16x32_bf16
    return __builtin_amdgcn_wmma_f32_16x16x32_bf16(false, a, false, b, (short)0, c, false, false);
}

// A fragment (16x32 bf16) from an LDS tile, ISA 7.12.2 layout:
// elem j<8 : K = kbase + 8*hi + j ; elem j>=8 : K = kbase + 16 + 8*hi + (j-8)
static __device__ __forceinline__ v16bf ldsAfrag(const __bf16* a, int r, int kbase, int hi, int stride) {
    const __bf16* pa = a + r * stride + kbase + hi * 8;
    V16 f;
    f.u[0] = *reinterpret_cast<const uint4*>(pa);
    f.u[1] = *reinterpret_cast<const uint4*>(pa + 16);
    return f.v;
}

// A fragment formed as elementwise product of two LDS tiles (v = sf * pos)
static __device__ __forceinline__ v16bf ldsAfragMul(const __bf16* a, const __bf16* b,
                                                    int r, int kbase, int hi, int stride) {
    const __bf16* pa = a + r * stride + kbase + hi * 8;
    const __bf16* pb = b + r * stride + kbase + hi * 8;
    H8 a0, a1, b0, b1;
    a0.u = *reinterpret_cast<const uint4*>(pa);
    a1.u = *reinterpret_cast<const uint4*>(pa + 16);
    b0.u = *reinterpret_cast<const uint4*>(pb);
    b1.u = *reinterpret_cast<const uint4*>(pb + 16);
    V16 f;
#pragma unroll
    for (int j = 0; j < 8; ++j) f.v[j]     = (__bf16)((float)a0.h[j] * (float)b0.h[j]);
#pragma unroll
    for (int j = 0; j < 8; ++j) f.v[8 + j] = (__bf16)((float)a1.h[j] * (float)b1.h[j]);
    return f.v;
}

// B fragment (32x16 bf16) preswizzled in workspace: 16 contiguous halves per lane
static __device__ __forceinline__ v16bf gblBfrag(const __bf16* base, int frag, int lane) {
    const __bf16* p = base + (long)frag * 512 + lane * 16;
    V16 f;
    f.u[0] = *reinterpret_cast<const uint4*>(p);
    f.u[1] = *reinterpret_cast<const uint4*>(p + 8);
    return f.v;
}

// ---- prologue: pack f32 weights into bf16 B-fragment layout in workspace ----
// B[k][n] = W[n][k]; fragment elem j for lane l: K = kt*32 + (l>=16?8:0) + (j&7) + (j>=8?16:0), N = nt*16 + (l&15)
__global__ void pack_weights_kernel(const float* __restrict__ c1w1, const float* __restrict__ c1w2,
                                    const float* __restrict__ c2w1, const float* __restrict__ c2w2,
                                    const float* __restrict__ gw1, __bf16* __restrict__ ws) {
    int id = blockIdx.x * 256 + threadIdx.x;
    if (id >= WS_HALVES) return;
    const float* W; int Ksz, ntiles, rel;
    if (id < OFF_C1W2)      { W = c1w1; Ksz = 256; ntiles = 4;  rel = id - OFF_C1W1; }
    else if (id < OFF_C2W1) { W = c1w2; Ksz = 64;  ntiles = 16; rel = id - OFF_C1W2; }
    else if (id < OFF_C2W2) { W = c2w1; Ksz = 256; ntiles = 4;  rel = id - OFF_C2W1; }
    else if (id < OFF_GW1)  { W = c2w2; Ksz = 64;  ntiles = 16; rel = id - OFF_C2W2; }
    else                    { W = gw1;  Ksz = 512; ntiles = 16; rel = id - OFF_GW1;  }
    int frag = rel >> 9;
    int lane = (rel >> 4) & 31;
    int j    = rel & 15;
    int kt = frag / ntiles, nt = frag % ntiles;
    int n = nt * 16 + (lane & 15);
    int k = kt * 32 + ((lane & 16) ? 8 : 0) + (j & 7) + ((j & 8) ? 16 : 0);
    ws[id] = (__bf16)W[n * Ksz + k];
}

// ---- fused main kernel: 32 tokens per block, 128 threads (4 waves) ----
__global__ void __launch_bounds__(128)
semantic_align_fused_kernel(const float* __restrict__ feat, const float* __restrict__ qbox,
                            const float* __restrict__ qf,   const float* __restrict__ pos,
                            const __bf16* __restrict__ wfr,
                            const float* __restrict__ c1b1, const float* __restrict__ c1b2,
                            const float* __restrict__ c1g,  const float* __restrict__ c1be,
                            const float* __restrict__ c2b1, const float* __restrict__ c2b2,
                            const float* __restrict__ c2g,  const float* __restrict__ c2be,
                            const float* __restrict__ gb1,  const float* __restrict__ gw2,
                            const float* __restrict__ gb2,  float* __restrict__ out) {
    __shared__ __bf16 sSF[TILE_M * LSTR];   // gathered start feature -> later pooled accumulator
    __shared__ __bf16 sEF[TILE_M * LSTR];   // gathered end feature   -> later gate hidden Hg
    __shared__ __bf16 sPS[TILE_M * LSTR];   // pos_feature            -> later LN input U
    __shared__ __bf16 sH[2][TILE_M * HSTR]; // contrast hidden (32x64 each)
    __shared__ int sIdxS[TILE_M], sIdxE[TILE_M];

    const int tid  = threadIdx.x;
    const int lane = tid & 31;
    const int wave = tid >> 5;
    const int hi   = lane >> 4;
    const int lr   = lane & 15;
    const long tok0 = (long)blockIdx.x * TILE_M;

    // S0: ROI indices per token row
    if (tid < TILE_M) {
        long t = tok0 + tid;
        float cx = qbox[t * 2 + 0], w = qbox[t * 2 + 1];
        float s = rintf(fminf(fmaxf((cx - 0.5f * w) * (float)LFEAT, 0.0f), (float)(LFEAT - 1)));
        float e = rintf(fminf(fmaxf((cx + 0.5f * w) * (float)LFEAT, 0.0f), (float)(LFEAT - 1)));
        sIdxS[tid] = (int)s;
        sIdxE[tid] = (int)e;
    }
    __syncthreads();

    // S1: stage sf (gather), ef (gather), pos as bf16 in LDS
    for (int it = tid; it < TILE_M * 64; it += 128) {   // 64 float4 chunks per row
        int r = it >> 6, c4 = it & 63;
        long t = tok0 + r;
        long b = t >> 12;   // / L1Q
        const float4 pv = *reinterpret_cast<const float4*>(pos + t * DIMC + c4 * 4);
        const float4 sv = *reinterpret_cast<const float4*>(feat + ((b * LFEAT + sIdxS[r]) * (long)DIMC) + c4 * 4);
        const float4 ev = *reinterpret_cast<const float4*>(feat + ((b * LFEAT + sIdxE[r]) * (long)DIMC) + c4 * 4);
        int o = r * LSTR + c4 * 4;
        sPS[o + 0] = (__bf16)pv.x; sPS[o + 1] = (__bf16)pv.y; sPS[o + 2] = (__bf16)pv.z; sPS[o + 3] = (__bf16)pv.w;
        sSF[o + 0] = (__bf16)sv.x; sSF[o + 1] = (__bf16)sv.y; sSF[o + 2] = (__bf16)sv.z; sSF[o + 3] = (__bf16)sv.w;
        sEF[o + 0] = (__bf16)ev.x; sEF[o + 1] = (__bf16)ev.y; sEF[o + 2] = (__bf16)ev.z; sEF[o + 3] = (__bf16)ev.w;
    }
    __syncthreads();

    // S2: contrast GEMM1 (v = sf*pos -> 256x64), both contrasts; H = relu(.+b1)
    for (int c = 0; c < 2; ++c) {
        const __bf16* sV1 = c ? sEF : sSF;
        const __bf16* wb  = wfr + (c ? OFF_C2W1 : OFF_C1W1);
        const float*  b1  = c ? c2b1 : c1b1;
        for (int tile = wave; tile < 2 * 4; tile += 4) {
            int mt = tile >> 2, nt = tile & 3;
            v8f acc = {};
#pragma unroll
            for (int kt = 0; kt < 8; ++kt) {
                v16bf a  = ldsAfragMul(sV1, sPS, mt * 16 + lr, kt * 32, hi, LSTR);
                v16bf bf = gblBfrag(wb, kt * 4 + nt, lane);
                acc = wmma_bf16(a, bf, acc);
            }
            float bias = b1[nt * 16 + lr];
#pragma unroll
            for (int j = 0; j < 8; ++j) {
                float h = fmaxf(acc[j] + bias, 0.0f);
                sH[c][(mt * 16 + j + hi * 8) * HSTR + nt * 16 + lr] = (__bf16)h;
            }
        }
    }
    __syncthreads();

    // S3: per contrast: GEMM2 (64->256) -> U (reuse sPS) -> LayerNorm -> sigmoid*v1 -> pooled into sSF
    for (int c = 0; c < 2; ++c) {
        const __bf16* wb = wfr + (c ? OFF_C2W2 : OFF_C1W2);
        const float*  b2 = c ? c2b2 : c1b2;
        const float*  gg = c ? c2g  : c1g;
        const float*  bb = c ? c2be : c1be;
        for (int tile = wave; tile < 2 * 16; tile += 4) {
            int mt = tile >> 4, nt = tile & 15;
            v8f acc = {};
#pragma unroll
            for (int kt = 0; kt < 2; ++kt) {
                v16bf a  = ldsAfrag(&sH[c][0], mt * 16 + lr, kt * 32, hi, HSTR);
                v16bf bf = gblBfrag(wb, kt * 16 + nt, lane);
                acc = wmma_bf16(a, bf, acc);
            }
            float bias = b2[nt * 16 + lr];
#pragma unroll
            for (int j = 0; j < 8; ++j)
                sPS[(mt * 16 + j + hi * 8) * LSTR + nt * 16 + lr] = (__bf16)(acc[j] + bias);
        }
        __syncthreads();
        {   // LayerNorm + sigmoid gate; 4 threads per row
            int r = tid >> 2, q = tid & 3;
            const __bf16* urow = sPS + r * LSTR + q * 64;
            float s = 0.f, ss = 0.f;
#pragma unroll 8
            for (int cc = 0; cc < 64; ++cc) { float u = (float)urow[cc]; s += u; ss += u * u; }
            s  += __shfl_xor(s, 1, 4);  s  += __shfl_xor(s, 2, 4);
            ss += __shfl_xor(ss, 1, 4); ss += __shfl_xor(ss, 2, 4);
            float mu   = s * (1.0f / 256.0f);
            float var  = ss * (1.0f / 256.0f) - mu * mu;
            float rstd = rsqrtf(var + 1e-6f);
            const __bf16* v1row = (c ? sEF : sSF) + r * LSTR + q * 64;
            __bf16* prow = sSF + r * LSTR + q * 64;
#pragma unroll 4
            for (int cc = 0; cc < 64; ++cc) {
                int col = q * 64 + cc;
                float u   = (float)urow[cc];
                float ln  = (u - mu) * rstd * gg[col] + bb[col];
                float sg  = 1.0f / (1.0f + __expf(-ln));
                float ctr = 0.5f * sg * (float)v1row[cc];
                prow[cc] = (c == 0) ? (__bf16)ctr : (__bf16)((float)prow[cc] + ctr);
            }
        }
        __syncthreads();
    }

    // S4: gate GEMM (x=[pooled,qf], 512 -> 256), Hg = relu(.+gb1) into sEF
    for (int tile = wave; tile < 2 * 16; tile += 4) {
        int mt = tile >> 4, nt = tile & 15;
        v8f acc = {};
#pragma unroll
        for (int kt = 0; kt < 16; ++kt) {
            v16bf a;
            if (kt < 8) {
                a = ldsAfrag(sSF, mt * 16 + lr, kt * 32, hi, LSTR);
            } else {
                long t = tok0 + mt * 16 + lr;
                const float* qp = qf + t * DIMC + (kt - 8) * 32 + hi * 8;
                float4 q0 = *reinterpret_cast<const float4*>(qp);
                float4 q1 = *reinterpret_cast<const float4*>(qp + 4);
                float4 q2 = *reinterpret_cast<const float4*>(qp + 16);
                float4 q3 = *reinterpret_cast<const float4*>(qp + 20);
                V16 f;
                f.v[0]  = (__bf16)q0.x; f.v[1]  = (__bf16)q0.y; f.v[2]  = (__bf16)q0.z; f.v[3]  = (__bf16)q0.w;
                f.v[4]  = (__bf16)q1.x; f.v[5]  = (__bf16)q1.y; f.v[6]  = (__bf16)q1.z; f.v[7]  = (__bf16)q1.w;
                f.v[8]  = (__bf16)q2.x; f.v[9]  = (__bf16)q2.y; f.v[10] = (__bf16)q2.z; f.v[11] = (__bf16)q2.w;
                f.v[12] = (__bf16)q3.x; f.v[13] = (__bf16)q3.y; f.v[14] = (__bf16)q3.z; f.v[15] = (__bf16)q3.w;
                a = f.v;
            }
            v16bf bf = gblBfrag(wfr + OFF_GW1, kt * 16 + nt, lane);
            acc = wmma_bf16(a, bf, acc);
        }
        float bias = gb1[nt * 16 + lr];
#pragma unroll
        for (int j = 0; j < 8; ++j) {
            float h = fmaxf(acc[j] + bias, 0.0f);
            sEF[(mt * 16 + j + hi * 8) * LSTR + nt * 16 + lr] = (__bf16)h;
        }
    }
    __syncthreads();

    // S5: gate logits (256->2), softmax, blend, write transposed [L1,B,C] output
    {
        int r = tid >> 2, q = tid & 3;
        long t = tok0 + r;
        long b = t >> 12, n = t & (L1Q - 1);
        const __bf16* hrow = sEF + r * LSTR + q * 64;
        float z0 = 0.f, z1 = 0.f;
#pragma unroll 8
        for (int cc = 0; cc < 64; ++cc) {
            int col = q * 64 + cc;
            float h = (float)hrow[cc];
            z0 += h * gw2[col];
            z1 += h * gw2[256 + col];
        }
        z0 += __shfl_xor(z0, 1, 4); z0 += __shfl_xor(z0, 2, 4);
        z1 += __shfl_xor(z1, 1, 4); z1 += __shfl_xor(z1, 2, 4);
        z0 += gb2[0]; z1 += gb2[1];
        float m  = fmaxf(z0, z1);
        float e0 = __expf(z0 - m), e1 = __expf(z1 - m);
        float g0 = e0 / (e0 + e1), g1 = 1.0f - g0;
        const __bf16* prow = sSF + r * LSTR + q * 64;
        const float*  qrow = qf + t * DIMC + q * 64;
        float* orow = out + (n * BATCH + b) * DIMC + q * 64;
#pragma unroll 4
        for (int cc = 0; cc < 64; ++cc)
            orow[cc] = g0 * (float)prow[cc] + g1 * qrow[cc];
    }
}

extern "C" void kernel_launch(void* const* d_in, const int* in_sizes, int n_in,
                              void* d_out, int out_size, void* d_ws, size_t ws_size,
                              hipStream_t stream) {
    const float* feat = (const float*)d_in[0];
    const float* qbox = (const float*)d_in[1];
    const float* qf   = (const float*)d_in[2];
    const float* pos  = (const float*)d_in[3];
    const float* c1w1 = (const float*)d_in[4];
    const float* c1b1 = (const float*)d_in[5];
    const float* c1w2 = (const float*)d_in[6];
    const float* c1b2 = (const float*)d_in[7];
    const float* c1g  = (const float*)d_in[8];
    const float* c1be = (const float*)d_in[9];
    const float* c2w1 = (const float*)d_in[10];
    const float* c2b1 = (const float*)d_in[11];
    const float* c2w2 = (const float*)d_in[12];
    const float* c2b2 = (const float*)d_in[13];
    const float* c2g  = (const float*)d_in[14];
    const float* c2be = (const float*)d_in[15];
    const float* gw1  = (const float*)d_in[16];
    const float* gb1  = (const float*)d_in[17];
    const float* gw2  = (const float*)d_in[18];
    const float* gb2  = (const float*)d_in[19];
    __bf16* wsb = (__bf16*)d_ws;
    float* out  = (float*)d_out;

    pack_weights_kernel<<<(WS_HALVES + 255) / 256, 256, 0, stream>>>(c1w1, c1w2, c2w1, c2w2, gw1, wsb);
    semantic_align_fused_kernel<<<NT_TOTAL / TILE_M, 128, 0, stream>>>(
        feat, qbox, qf, pos, wsb,
        c1b1, c1b2, c1g, c1be, c2b1, c2b2, c2g, c2be,
        gb1, gw2, gb2, out);
}
